// SA_62508954026085
// MI455X (gfx1250) — matrix operand order repaired and checked by reading, not compiled, not run on hardware
//
#include <hip/hip_runtime.h>
#include <hip/hip_bf16.h>

#define B_ 32
#define T_ 4096
#define D_ 256
#define S_ 8
#define ITERS_ 3
#define SCALE_ 0.0625f   /* 256^-0.5 */
#define EPS_ 1e-8f

typedef __bf16 bf16_t;
typedef bf16_t bf16x8  __attribute__((ext_vector_type(8)));
typedef bf16_t bf16x16 __attribute__((ext_vector_type(16)));
typedef float  f32x8   __attribute__((ext_vector_type(8)));

#if defined(__has_builtin)
#if __has_builtin(__builtin_amdgcn_global_load_async_to_lds_b128) && \
    __has_builtin(__builtin_amdgcn_s_wait_asynccnt)
#define HAVE_ASYNC_LDS 1
#endif
#endif

#if defined(HAVE_ASYNC_LDS)
// Param types per hipcc diagnostic: gcc-vector int4, AS1 (global) / AS3 (LDS)
typedef int v4i_ __attribute__((vector_size(16)));
typedef __attribute__((address_space(1))) v4i_* as1_v4i;
typedef __attribute__((address_space(3))) v4i_* as3_v4i;
#endif

// ---- WMMA fragment loaders (CDNA5 wave32 layouts, CDNA5 ISA) -------------
// A 16x32 bf16: lane&15 = M; lane<16 holds K {0..7,16..23}, lane>=16 {8..15,24..31}
__device__ __forceinline__ bf16x16 frag_a(const bf16_t* base, int stride, int row,
                                          int k0, int lane) {
  const int kh = (lane >> 4) << 3;
  bf16x8 lo = *reinterpret_cast<const bf16x8*>(base + (size_t)row * stride + k0 + kh);
  bf16x8 hi = *reinterpret_cast<const bf16x8*>(base + (size_t)row * stride + k0 + 16 + kh);
  return __builtin_shufflevector(lo, hi, 0,1,2,3,4,5,6,7,8,9,10,11,12,13,14,15);
}
// B 32x16 bf16 stored [N][K]: lane&15 = N; lane<16 holds K {0..7,16..23}, lane>=16 {8..15,24..31}
__device__ __forceinline__ bf16x16 frag_b(const bf16_t* base, int stride, int nrow,
                                          int k0, int lane) {
  const int kb = (lane >> 4) << 3;
  bf16x8 lo = *reinterpret_cast<const bf16x8*>(base + (size_t)nrow * stride + k0 + kb);
  bf16x8 hi = *reinterpret_cast<const bf16x8*>(base + (size_t)nrow * stride + k0 + 16 + kb);
  return __builtin_shufflevector(lo, hi, 0,1,2,3,4,5,6,7,8,9,10,11,12,13,14,15);
}
#define WMMA_BF16(a, b, c) \
  __builtin_amdgcn_wmma_f32_16x16x32_bf16(false, (a), false, (b), (short)0, (c), false, false)

// ---- slots = mu + exp(logsigma) * noise -----------------------------------
__global__ void SA_init_slots(const float* __restrict__ mu, const float* __restrict__ ls,
                              const float* __restrict__ noise, float* __restrict__ slots) {
  int i = blockIdx.x * blockDim.x + threadIdx.x;
  if (i < B_ * S_ * D_) {
    int d = i & (D_ - 1);
    slots[i] = mu[d] + __expf(ls[d]) * noise[i];
  }
}

// ---- keys/values projection -----------------------------------------------
// keys stored [B*T][D] (row major), vals stored TRANSPOSED [D][B*T].
__global__ __launch_bounds__(512) void SA_project_kv(
    const float* __restrict__ x, const float* __restrict__ Wk,
    const float* __restrict__ Wv, bf16_t* __restrict__ keys, bf16_t* __restrict__ valsT) {
  __shared__ bf16_t aT[64 * D_];       // 64 input rows bf16; reused to stage keys out
  __shared__ bf16_t wkT[D_ * 32];      // weight chunk, transposed [N][K]
  __shared__ bf16_t wvT[D_ * 32];
  const int tid = threadIdx.x, lane = tid & 31, wave = tid >> 5;
  const int row0 = blockIdx.x * 64;
  for (int i = tid; i < 64 * D_; i += 512)
    aT[i] = (bf16_t)x[(size_t)row0 * D_ + i];
  f32x8 accK[4] = {}; f32x8 accV[4] = {};
  const int n0 = wave * 16;
  for (int kc = 0; kc < 8; ++kc) {
    const int k0 = kc * 32;
    __syncthreads();
    for (int i = tid; i < 32 * D_; i += 512) {
      int k = i >> 8, n = i & 255;
      wkT[n * 32 + k] = (bf16_t)Wk[(k0 + k) * D_ + n];
      wvT[n * 32 + k] = (bf16_t)Wv[(k0 + k) * D_ + n];
    }
    __syncthreads();
    bf16x16 bK = frag_b(wkT, 32, n0 + (lane & 15), 0, lane);
    bf16x16 bV = frag_b(wvT, 32, n0 + (lane & 15), 0, lane);
#pragma unroll
    for (int st = 0; st < 4; ++st) {
      bf16x16 a = frag_a(aT, D_, st * 16 + (lane & 15), k0, lane);
      accK[st] = WMMA_BF16(a, bK, accK[st]);
      accV[st] = WMMA_BF16(a, bV, accV[st]);
    }
  }
  const int n = n0 + (lane & 15);
  const int mh = (lane >> 4) * 8;
  // values: lane holds 8 consecutive output rows at column n -> one b128 store
  // per strip into the transposed layout valsT[n][B*T + m].
#pragma unroll
  for (int st = 0; st < 4; ++st) {
    bf16x8 pv;
#pragma unroll
    for (int r = 0; r < 8; ++r) pv[r] = (bf16_t)accV[st][r];
    *reinterpret_cast<bf16x8*>(&valsT[(size_t)n * (B_ * T_) + row0 + st * 16 + mh]) = pv;
  }
  // keys: stage through LDS, then coalesced b128 stores
  __syncthreads();                        // aT frag reads done in all waves
#pragma unroll
  for (int st = 0; st < 4; ++st)
#pragma unroll
    for (int r = 0; r < 8; ++r)
      aT[(st * 16 + mh + r) * D_ + n] = (bf16_t)accK[st][r];
  __syncthreads();
  for (int i = tid; i < (64 * D_) / 8; i += 512)
    *reinterpret_cast<bf16x8*>(&keys[(size_t)row0 * D_ + i * 8]) =
        *reinterpret_cast<const bf16x8*>(&aT[i * 8]);
}

// ---- q = LayerNorm(slots) @ Wq, bf16 padded to 16 rows --------------------
__global__ __launch_bounds__(256) void SA_qproj(
    const float* __restrict__ slots, const float* __restrict__ g,
    const float* __restrict__ bb, const float* __restrict__ Wq,
    bf16_t* __restrict__ qout) {
  __shared__ float sn[S_][D_];
  const int b = blockIdx.x, tid = threadIdx.x;
  const int lane = tid & 31, wave = tid >> 5;   // 8 waves: one slot row each
  const float* row = slots + ((size_t)b * S_ + wave) * D_;
  float v[8]; float s1 = 0.f, s2 = 0.f;
#pragma unroll
  for (int i = 0; i < 8; ++i) { v[i] = row[lane + 32 * i]; s1 += v[i]; s2 += v[i] * v[i]; }
#pragma unroll
  for (int off = 16; off > 0; off >>= 1) {
    s1 += __shfl_xor(s1, off, 32);
    s2 += __shfl_xor(s2, off, 32);
  }
  const float mu = s1 * (1.f / D_);
  const float rstd = rsqrtf(s2 * (1.f / D_) - mu * mu + 1e-5f);
#pragma unroll
  for (int i = 0; i < 8; ++i) {
    int d = lane + 32 * i;
    sn[wave][d] = (v[i] - mu) * rstd * g[d] + bb[d];
  }
  __syncthreads();
  const int j = tid;
#pragma unroll
  for (int s = 0; s < S_; ++s) {
    float acc = 0.f;
    for (int k = 0; k < D_; ++k) acc += sn[s][k] * Wq[k * D_ + j];
    qout[((size_t)b * 16 + s) * D_ + j] = (bf16_t)acc;
  }
#pragma unroll
  for (int s = S_; s < 16; ++s) qout[((size_t)b * 16 + s) * D_ + j] = (bf16_t)0.f;
}

__global__ void SA_zero(float* __restrict__ U, float* __restrict__ W) {
  int i = blockIdx.x * blockDim.x + threadIdx.x;
  if (i < B_ * S_ * D_) U[i] = 0.f;
  if (i < B_ * S_) W[i] = 0.f;
}

// ---- streaming attention: dots -> slot softmax -> accumulate W, U ---------
__global__ __launch_bounds__(512) void SA_attn(
    const bf16_t* __restrict__ qin, const bf16_t* __restrict__ keys,
    const bf16_t* __restrict__ valsT, float* __restrict__ U, float* __restrict__ W) {
  __shared__ bf16_t qT[16 * D_];        // padded q tile for this batch
  __shared__ bf16_t attnT[16 * 256];    // a[s][t_local], rows 8..15 zero
  __shared__ float  Wred[S_];
  const int tid = threadIdx.x, lane = tid & 31, wave = tid >> 5;
  const int b = blockIdx.x >> 4;                // 16 t-chunks per batch
  const int t_base = (blockIdx.x & 15) * 256;
  {
    const bf16_t* src = qin + (size_t)b * 16 * D_;
#if defined(HAVE_ASYNC_LDS)
    // 8 KB tile: one b128 async copy per thread, tracked by ASYNCcnt
    __builtin_amdgcn_global_load_async_to_lds_b128(
        (as1_v4i)((const char*)src + tid * 16),
        (as3_v4i)((char*)qT + tid * 16), 0, 0);
    __builtin_amdgcn_s_wait_asynccnt(0);
#else
    for (int i = tid; i < 16 * D_; i += 512) qT[i] = src[i];
#endif
  }
  if (tid < S_) Wred[tid] = 0.f;
  __syncthreads();

  // phase 1: dots tile (M=16 slots-pad, N=16 t per wave, K=256)
  f32x8 acc = {};
  const int tloc = wave * 16 + (lane & 15);
  const bf16_t* krow = keys + ((size_t)b * T_ + t_base) * D_;
#pragma unroll
  for (int kc = 0; kc < 8; ++kc) {
    bf16x16 a  = frag_a(qT, D_, lane & 15, kc * 32, lane);
    bf16x16 bt = frag_b(krow, D_, tloc, kc * 32, lane);   // keys are [t][d] = [N][K]
    acc = WMMA_BF16(a, bt, acc);
  }
  // lanes 0..15 hold all 8 real slot scores for t = t_base + tloc
  if (lane < 16) {
    float e[S_]; float m = -1e30f;
#pragma unroll
    for (int r = 0; r < S_; ++r) { e[r] = acc[r] * SCALE_; m = fmaxf(m, e[r]); }
    float s = 0.f;
#pragma unroll
    for (int r = 0; r < S_; ++r) { e[r] = __expf(e[r] - m); s += e[r]; }
    const float inv = 1.f / s;
#pragma unroll
    for (int r = 0; r < S_; ++r) {
      float av = e[r] * inv + EPS_;
      attnT[r * 256 + tloc] = (bf16_t)av;
      atomicAdd(&Wred[r], av);            // running per-slot T-sum
    }
  } else {
#pragma unroll
    for (int r = 0; r < S_; ++r)          // pad rows 8..15 -> zero weight
      attnT[(r + 8) * 256 + wave * 16 + (lane & 15)] = (bf16_t)0.f;
  }
  __syncthreads();
  if (tid < S_) atomicAdd(&W[b * S_ + tid], Wred[tid]);

  // phase 2: U_partial = attn(16x256) @ V(256x256); valsT is already [N=d][K=t]
  f32x8 accU = {};
  const int n0 = wave * 16;
  const bf16_t* vbase = valsT + (size_t)b * T_ + t_base;
#pragma unroll
  for (int kc = 0; kc < 8; ++kc) {
    bf16x16 a  = frag_a(attnT, 256, lane & 15, kc * 32, lane);
    bf16x16 bt = frag_b(vbase, B_ * T_, n0 + (lane & 15), kc * 32, lane);
    accU = WMMA_BF16(a, bt, accU);
  }
  if (lane < 16) {
#pragma unroll
    for (int r = 0; r < S_; ++r)
      atomicAdd(&U[((size_t)b * S_ + r) * D_ + n0 + lane], accU[r]);
  }
}

// ---- GRU + pre-norm residual MLP: one (b,s) row per block -----------------
__global__ __launch_bounds__(256) void SA_update(
    float* __restrict__ slots, const float* __restrict__ U, const float* __restrict__ W,
    const float* __restrict__ wih, const float* __restrict__ whh,
    const float* __restrict__ bih, const float* __restrict__ bhh,
    const float* __restrict__ ng, const float* __restrict__ nb,
    const float* __restrict__ w1, const float* __restrict__ b1,
    const float* __restrict__ w2, const float* __restrict__ b2) {
  __shared__ float upd[D_];
  __shared__ float hp[D_];
  __shared__ float xn[D_];
  __shared__ float hbuf[4 * D_];
  __shared__ float red1[8], red2[8];
  const int bs = blockIdx.x;                 // b * 8 + s
  const int j = threadIdx.x, lane = j & 31, wave = j >> 5;
  upd[j] = U[(size_t)bs * D_ + j] / W[bs];
  hp[j]  = slots[(size_t)bs * D_ + j];
  __syncthreads();
  float ir = bih[j], iz = bih[D_ + j], in_ = bih[2 * D_ + j];
  float hr = bhh[j], hz = bhh[D_ + j], hn = bhh[2 * D_ + j];
  for (int k = 0; k < D_; ++k) {
    const float u = upd[k], h = hp[k];
    ir += u * wih[k * 3 * D_ + j];           hr += h * whh[k * 3 * D_ + j];
    iz += u * wih[k * 3 * D_ + D_ + j];      hz += h * whh[k * 3 * D_ + D_ + j];
    in_ += u * wih[k * 3 * D_ + 2 * D_ + j]; hn += h * whh[k * 3 * D_ + 2 * D_ + j];
  }
  const float r = 1.f / (1.f + __expf(-(ir + hr)));
  const float z = 1.f / (1.f + __expf(-(iz + hz)));
  const float nn = tanhf(in_ + r * hn);
  const float hnew = (1.f - z) * nn + z * hp[j];
  // block-wide LayerNorm over the 256 hnew values
  float s1 = hnew, s2 = hnew * hnew;
#pragma unroll
  for (int off = 16; off > 0; off >>= 1) {
    s1 += __shfl_xor(s1, off, 32);
    s2 += __shfl_xor(s2, off, 32);
  }
  if (lane == 0) { red1[wave] = s1; red2[wave] = s2; }
  __syncthreads();
  float t1 = 0.f, t2 = 0.f;
#pragma unroll
  for (int wv = 0; wv < 8; ++wv) { t1 += red1[wv]; t2 += red2[wv]; }
  const float mu = t1 * (1.f / D_);
  const float rstd = rsqrtf(t2 * (1.f / D_) - mu * mu + 1e-5f);
  xn[j] = (hnew - mu) * rstd * ng[j] + nb[j];
  __syncthreads();
#pragma unroll
  for (int c4 = 0; c4 < 4; ++c4) {
    const int c = c4 * D_ + j;
    float acc = b1[c];
    for (int k = 0; k < D_; ++k) acc += xn[k] * w1[k * 4 * D_ + c];
    hbuf[c] = fmaxf(acc, 0.f);
  }
  __syncthreads();
  float acc = b2[j];
  for (int k = 0; k < 4 * D_; ++k) acc += hbuf[k] * w2[k * D_ + j];
  slots[(size_t)bs * D_ + j] = hnew + acc;
}

extern "C" void kernel_launch(void* const* d_in, const int* in_sizes, int n_in,
                              void* d_out, int out_size, void* d_ws, size_t ws_size,
                              hipStream_t stream) {
  const float* x    = (const float*)d_in[0];
  const float* mu   = (const float*)d_in[1];
  const float* ls   = (const float*)d_in[2];
  const float* nz   = (const float*)d_in[3];
  const float* Wq   = (const float*)d_in[4];
  const float* Wk   = (const float*)d_in[5];
  const float* Wv   = (const float*)d_in[6];
  const float* ng0  = (const float*)d_in[7];
  const float* nb0  = (const float*)d_in[8];
  const float* wih  = (const float*)d_in[9];
  const float* whh  = (const float*)d_in[10];
  const float* bih  = (const float*)d_in[11];
  const float* bhh  = (const float*)d_in[12];
  const float* mng  = (const float*)d_in[13];
  const float* mnb  = (const float*)d_in[14];
  const float* w1   = (const float*)d_in[15];
  const float* b1   = (const float*)d_in[16];
  const float* w2   = (const float*)d_in[17];
  const float* b2   = (const float*)d_in[18];

  char* ws = (char*)d_ws;
  const size_t kvBytes = (size_t)B_ * T_ * D_ * sizeof(bf16_t);   // 64 MB each
  bf16_t* keys  = (bf16_t*)ws;
  bf16_t* valsT = (bf16_t*)(ws + kvBytes);
  bf16_t* qbuf  = (bf16_t*)(ws + 2 * kvBytes);
  float*  U     = (float*)(ws + 2 * kvBytes + (size_t)B_ * 16 * D_ * sizeof(bf16_t));
  float*  W     = (float*)((char*)U + (size_t)B_ * S_ * D_ * sizeof(float));
  float*  slots = (float*)d_out;

  SA_init_slots<<<(B_ * S_ * D_ + 255) / 256, 256, 0, stream>>>(mu, ls, nz, slots);
  SA_project_kv<<<(B_ * T_) / 64, 512, 0, stream>>>(x, Wk, Wv, keys, valsT);
  for (int it = 0; it < ITERS_; ++it) {
    SA_qproj<<<B_, 256, 0, stream>>>(slots, ng0, nb0, Wq, qbuf);
    SA_zero<<<(B_ * S_ * D_ + 255) / 256, 256, 0, stream>>>(U, W);
    SA_attn<<<B_ * (T_ / 256), 512, 0, stream>>>(qbuf, keys, valsT, U, W);
    SA_update<<<B_ * S_, 256, 0, stream>>>(slots, U, W, wih, whh, bih, bhh,
                                           mng, mnb, w1, b1, w2, b2);
  }
}
